// DirectHBHead_35673998360869
// MI455X (gfx1250) — compile-verified, matrix operand order checked by reading, not run on hardware
//
#include <hip/hip_runtime.h>
#include <hip/hip_bf16.h>

typedef __bf16 bf16_t;
typedef __attribute__((ext_vector_type(16))) __bf16 v16bf;
typedef __attribute__((ext_vector_type(8)))  float  v8f;

#define EPSN 1e-12f
#define SCALE_S 20.0f

__device__ inline unsigned bfbits(float f) {   // RNE f32 -> bf16 bits
  unsigned u; __builtin_memcpy(&u, &f, 4);
  return (u + 0x7FFFu + ((u >> 16) & 1u)) >> 16;
}
__device__ inline bf16_t f2bf(float f) {
  unsigned short s = (unsigned short)bfbits(f);
  bf16_t b; __builtin_memcpy(&b, &s, 2);
  return b;
}

// Fragment loader for v_wmma_f32_16x16x32_bf16.
// Per ISA layout: lane (l&15) = row/col index; lanes<16 hold K = {0..7,16..23},
// lanes>=16 hold K = {8..15,24..31}. Caller passes p = rowbase + (l>>4)*8.
// Two 16B vector loads (all call sites are 16B aligned).
__device__ inline v16bf frag_bf(const bf16_t* p) {
  union { uint4 q[2]; v16bf v; } U;
  U.q[0] = *(const uint4*)(p);
  U.q[1] = *(const uint4*)(p + 16);
  return U.v;
}
__device__ inline v8f zero8() {
  v8f z;
#pragma unroll
  for (int i = 0; i < 8; ++i) z[i] = 0.0f;
  return z;
}
__device__ inline v8f wmma_bf16(v16bf a, v16bf b, v8f c) {
  return __builtin_amdgcn_wmma_f32_16x16x32_bf16(false, a, false, b, (short)0, c, false, false);
}

__global__ void zero_u32(unsigned* p, int n) {
  int i = blockIdx.x * blockDim.x + threadIdx.x;
  if (i < n) p[i] = 0u;
}

// packed f32 -> bf16 conversion: each thread converts 2 floats -> one u32
__global__ void cvt_f32_bf16(const float* __restrict__ src, unsigned* __restrict__ dst, int n2) {
  int i = blockIdx.x * blockDim.x + threadIdx.x;
  if (i < n2) {
    unsigned lo = bfbits(src[2 * i]);
    unsigned hi = bfbits(src[2 * i + 1]);
    dst[i] = (hi << 16) | lo;
  }
}

// vbar[v] = mean_p vf[v,1+p,:]  (768);  tbar[t] = sum_a tf[t,1+a,:]
__global__ void means_kernel(const float* __restrict__ vf, const float* __restrict__ tf,
                             float* __restrict__ vbar, float* __restrict__ tbar) {
  int bid = blockIdx.x;
  for (int c = threadIdx.x; c < 768; c += blockDim.x) {
    if (bid < 32) {
      int v = bid; float s = 0.f;
      for (int p = 0; p < 192; ++p) s += vf[((size_t)(v * 193 + 1 + p)) * 768 + c];
      vbar[v * 768 + c] = s * (1.0f / 192.0f);
    } else {
      int t = bid - 32; float s = 0.f;
      for (int a = 0; a < 30; ++a) s += tf[((size_t)(t * 31 + 1 + a)) * 768 + c];
      tbar[t * 768 + c] = s;
    }
  }
}

// Generic projection: 32-row slab, X(rows x 768, bf16) @ W^T(768->256, bf16) via WMMA.
// mode 0: visual_embed  1: textual_embed  2: patch_mean  3: att_mean
// mode 4: patch -> patchN(l2n bf16)[v][p][e] + patchT bf16 [v][e][p]
// mode 5: att   -> attN  (l2n bf16)[t][a(32)][e] + attT bf16 [t][e][a(32)]
__global__ void proj_kernel(const bf16_t* __restrict__ X, const bf16_t* __restrict__ W,
                            const float* __restrict__ bias, const int* __restrict__ att_nums,
                            float* __restrict__ outF, bf16_t* __restrict__ outN,
                            bf16_t* __restrict__ outT, int mode) {
  __shared__ float sOut[32][256];
  __shared__ size_t sRowOff[32];
  __shared__ float sRed[256];
  __shared__ float sInv[32];
  int tid = threadIdx.x, w = tid >> 5, lane = tid & 31;
  int rbase = blockIdx.x * 32;
  if (tid < 32) {
    int r = rbase + tid; size_t off;
    switch (mode) {
      case 0: off = (size_t)r * (193 * 768); break;
      case 1: off = (size_t)r * (31 * 768); break;
      case 4: off = ((size_t)(r / 192) * 193 + 1 + (r % 192)) * 768; break;
      case 5: off = ((size_t)(r / 30) * 31 + 1 + (r % 30)) * 768; break;
      default: off = (size_t)r * 768; break;
    }
    sRowOff[tid] = off;
  }
  __syncthreads();
  int mrow = lane & 15, kg = lane >> 4;
  for (int tile = w; tile < 32; tile += 8) {     // per-wave-uniform: EXEC all-ones at WMMA
    int mt = tile >> 4, nt = tile & 15;
    v8f acc = zero8();
    const bf16_t* xrow = X + sRowOff[mt * 16 + mrow] + kg * 8;
    const bf16_t* wrow = W + (size_t)(nt * 16 + mrow) * 768 + kg * 8;
    for (int k0 = 0; k0 < 768; k0 += 32)
      acc = wmma_bf16(frag_bf(xrow + k0), frag_bf(wrow + k0), acc);
    int dr0 = mt * 16 + kg * 8, col = nt * 16 + mrow;
#pragma unroll
    for (int r = 0; r < 8; ++r) sOut[dr0 + r][col] = acc[r];
  }
  __syncthreads();

  if (mode <= 2) {
    for (int idx = tid; idx < 32 * 256; idx += 256) {
      int r = idx >> 8, e = idx & 255;
      outF[(size_t)(rbase + r) * 256 + e] = sOut[r][e] + bias[e];
    }
    return;
  }
  if (mode == 3) {
    for (int idx = tid; idx < 32 * 256; idx += 256) {
      int r = idx >> 8, e = idx & 255;
      float an = (float)att_nums[r];
      outF[(size_t)r * 256 + e] = (sOut[r][e] + 30.0f * bias[e]) / an;
    }
    return;
  }
  // modes 4/5: add bias, then per-row l2 norm
  for (int idx = tid; idx < 32 * 256; idx += 256) {
    int r = idx >> 8, e = idx & 255;
    sOut[r][e] += bias[e];
  }
  __syncthreads();
  {
    int rr = tid >> 3, sub = tid & 7;
    float s = 0.f;
    for (int j = 0; j < 32; ++j) { float x = sOut[rr][sub * 32 + j]; s += x * x; }
    sRed[rr * 8 + sub] = s;
  }
  __syncthreads();
  if (tid < 32) {
    float ss = 0.f;
    for (int j = 0; j < 8; ++j) ss += sRed[tid * 8 + j];
    sInv[tid] = 1.0f / fmaxf(sqrtf(ss), EPSN);
  }
  __syncthreads();
  for (int idx = tid; idx < 32 * 256; idx += 256) {
    int r = idx >> 8, e = idx & 255;
    float val = sOut[r][e];
    int R = rbase + r;
    if (mode == 4) {
      int v = R / 192, p = R % 192;
      outN[((size_t)(v * 192 + p)) * 256 + e] = f2bf(val * sInv[r]);
      outT[((size_t)(v * 256 + e)) * 192 + p] = f2bf(val);
    } else {
      int t = R / 30, a = R % 30;
      outN[((size_t)(t * 32 + a)) * 256 + e] = f2bf(val * sInv[r]);
      outT[((size_t)(t * 256 + e)) * 32 + a] = f2bf(val);
    }
  }
}

// One workgroup per (v,t) pair. 256 threads = 8 waves.
__global__ void pair_kernel(const bf16_t* __restrict__ patchN, const bf16_t* __restrict__ patchT,
                            const bf16_t* __restrict__ attN, const bf16_t* __restrict__ attT,
                            const int* __restrict__ att_nums, float* __restrict__ outSim) {
  extern __shared__ char smem[];
  float*  sS     = (float*)smem;          // [32][192] relu(scores)*20
  float*  sAP    = sS + 32 * 192;         // [32][256] attend_patch
  float*  sAA    = sAP + 32 * 256;        // [16][256] attend_att p-block
  float*  sSap   = sAA + 16 * 256;        // [256]
  float*  sSaa   = sSap + 256;            // [256]
  float*  sRed   = sSaa + 256;            // [512]
  float*  sInv   = sRed + 512;            // [32]
  float*  sMax   = sInv + 32;             // [32]
  float*  sSum   = sMax + 32;             // [32]
  bf16_t* sAttn1 = (bf16_t*)(sSum + 32);  // [32][192]
  bf16_t* sAttn2 = sAttn1 + 32 * 192;     // [192][32]

  int tid = threadIdx.x, w = tid >> 5, lane = tid & 31;
  int v = blockIdx.x & 31, t = blockIdx.x >> 5;
  int an = att_nums[t];
  int mrow = lane & 15, kg = lane >> 4;

  // ---- GEMM1: S = relu(att_n[t] (32x256) @ patch_n[v]^T (256x192)) * 20
  for (int tile = w; tile < 24; tile += 8) {
    int mt = tile / 12, nt = tile % 12;
    v8f acc = zero8();
    const bf16_t* arow = attN + ((size_t)(t * 32 + mt * 16 + mrow)) * 256 + kg * 8;
    const bf16_t* brow = patchN + ((size_t)(v * 192 + nt * 16 + mrow)) * 256 + kg * 8;
    for (int k0 = 0; k0 < 256; k0 += 32)
      acc = wmma_bf16(frag_bf(arow + k0), frag_bf(brow + k0), acc);
    int dr0 = mt * 16 + kg * 8, col = nt * 16 + mrow;
#pragma unroll
    for (int r = 0; r < 8; ++r) sS[(dr0 + r) * 192 + col] = fmaxf(acc[r], 0.f) * SCALE_S;
  }
  __syncthreads();

  // ---- softmax over p (attn1), 8 threads per attribute row
  {
    int a = tid >> 3, sub = tid & 7, c0 = sub * 24;
    float m = 0.f;                                   // entries >= 0
    for (int j = 0; j < 24; ++j) m = fmaxf(m, sS[a * 192 + c0 + j]);
    sRed[a * 8 + sub] = m;
    __syncthreads();
    if (sub == 0) {
      float mm = 0.f;
      for (int j = 0; j < 8; ++j) mm = fmaxf(mm, sRed[a * 8 + j]);
      sMax[a] = mm;
    }
    __syncthreads();
    float s = 0.f;
    for (int j = 0; j < 24; ++j) s += __expf(sS[a * 192 + c0 + j] - sMax[a]);
    sRed[a * 8 + sub] = s;
    __syncthreads();
    if (sub == 0) {
      float ss = 0.f;
      for (int j = 0; j < 8; ++j) ss += sRed[a * 8 + j];
      sSum[a] = ss;
    }
    __syncthreads();
    float inv = 1.0f / sSum[a];
    for (int j = 0; j < 24; ++j)
      sAttn1[a * 192 + c0 + j] = f2bf(__expf(sS[a * 192 + c0 + j] - sMax[a]) * inv);
  }
  __syncthreads();

  // ---- softmax over a with padding mask (attn2); s2[p,a] == s1[a,p]
  if (tid < 192) {
    int p = tid;
    float m = -1e30f;
    for (int a = 0; a < an; ++a) m = fmaxf(m, sS[a * 192 + p]);
    float s = 0.f;
    for (int a = 0; a < an; ++a) s += __expf(sS[a * 192 + p] - m);
    float inv = 1.0f / s;
    for (int a = 0; a < 32; ++a)
      sAttn2[p * 32 + a] = (a < an) ? f2bf(__expf(sS[a * 192 + p] - m) * inv) : f2bf(0.f);
  }
  __syncthreads();

  // ---- GEMM2: attend_patch (32x256) = attn1 (32x192) @ patch[v] (192x256); B^T = patchT rows
  for (int tile = w; tile < 32; tile += 8) {
    int mt = tile >> 4, nt = tile & 15;
    v8f acc = zero8();
    const bf16_t* arow = sAttn1 + (mt * 16 + mrow) * 192 + kg * 8;
    const bf16_t* brow = patchT + ((size_t)(v * 256 + nt * 16 + mrow)) * 192 + kg * 8;
    for (int k0 = 0; k0 < 192; k0 += 32)
      acc = wmma_bf16(frag_bf(arow + k0), frag_bf(brow + k0), acc);
    int dr0 = mt * 16 + kg * 8, col = nt * 16 + mrow;
#pragma unroll
    for (int r = 0; r < 8; ++r) sAP[(dr0 + r) * 256 + col] = acc[r];
  }
  __syncthreads();

  // ---- S_ap[e] = sum_{a<an} l2n(attend_patch[a])[e]
  {
    int r = tid >> 3, sub = tid & 7;
    float s = 0.f;
    for (int j = 0; j < 32; ++j) { float x = sAP[r * 256 + sub * 32 + j]; s += x * x; }
    sRed[r * 8 + sub] = s;
    __syncthreads();
    if (tid < 32) {
      float ss = 0.f;
      for (int j = 0; j < 8; ++j) ss += sRed[tid * 8 + j];
      sInv[tid] = (tid < an) ? 1.0f / fmaxf(sqrtf(ss), EPSN) : 0.f;
    }
    __syncthreads();
    float acc = 0.f;
    for (int r2 = 0; r2 < an; ++r2) acc += sAP[r2 * 256 + tid] * sInv[r2];
    sSap[tid] = acc;
    sSaa[tid] = 0.f;
  }
  __syncthreads();

  // ---- GEMM3 in 16-row p-blocks: attend_att = attn2 (192x32) @ att[t] (32x256); K=32 -> 1 WMMA
  for (int pb = 0; pb < 12; ++pb) {
    for (int i = 0; i < 2; ++i) {
      int nt = w * 2 + i;                      // per-wave-uniform
      v8f acc = zero8();
      const bf16_t* arow = sAttn2 + (pb * 16 + mrow) * 32 + kg * 8;
      const bf16_t* brow = attT + ((size_t)(t * 256 + nt * 16 + mrow)) * 32 + kg * 8;
      acc = wmma_bf16(frag_bf(arow), frag_bf(brow), acc);
      int dr0 = kg * 8, col = nt * 16 + mrow;
#pragma unroll
      for (int r = 0; r < 8; ++r) sAA[(dr0 + r) * 256 + col] = acc[r];
    }
    __syncthreads();
    int r = tid >> 4, sub = tid & 15;
    float s = 0.f;
    for (int j = 0; j < 16; ++j) { float x = sAA[r * 256 + sub * 16 + j]; s += x * x; }
    sRed[r * 16 + sub] = s;
    __syncthreads();
    if (tid < 16) {
      float ss = 0.f;
      for (int j = 0; j < 16; ++j) ss += sRed[tid * 16 + j];
      sInv[tid] = 1.0f / fmaxf(sqrtf(ss), EPSN);
    }
    __syncthreads();
    float acc2 = 0.f;
    for (int r2 = 0; r2 < 16; ++r2) acc2 += sAA[r2 * 256 + tid] * sInv[r2];
    sSaa[tid] += acc2;
    __syncthreads();
  }

  // ---- final dot: local_similarity[t,v] = (S_aa . S_ap) / (30*an)
  sRed[tid] = sSap[tid] * sSaa[tid];
  __syncthreads();
  for (int s = 128; s > 0; s >>= 1) {
    if (tid < s) sRed[tid] += sRed[tid + s];
    __syncthreads();
  }
  if (tid == 0) outSim[t * 32 + v] = sRed[0] / (30.0f * (float)an);
}

extern "C" void kernel_launch(void* const* d_in, const int* in_sizes, int n_in,
                              void* d_out, int out_size, void* d_ws, size_t ws_size,
                              hipStream_t stream) {
  const float* vf = (const float*)d_in[0];
  const float* tf = (const float*)d_in[1];
  // d_in[2] = mask (consistent with att_nums; unused)
  const int* att_nums = (const int*)d_in[3];
  const float* W_vis = (const float*)d_in[4];
  const float* b_vis = (const float*)d_in[5];
  const float* W_txt = (const float*)d_in[6];
  const float* b_txt = (const float*)d_in[7];
  const float* W_patch = (const float*)d_in[8];
  const float* b_patch = (const float*)d_in[9];
  const float* W_att = (const float*)d_in[10];
  const float* b_att = (const float*)d_in[11];

  float* out = (float*)d_out;
  float* out_vis = out;
  float* out_txt = out + 32 * 256;
  float* out_pm  = out + 2 * 32 * 256;
  float* out_am  = out + 3 * 32 * 256;
  float* out_sim = out + 4 * 32 * 256;

  char* ws = (char*)d_ws;
  size_t o = 0;
  auto walloc = [&](size_t bytes) -> void* {
    void* p = ws + o;
    o += (bytes + 255) & ~(size_t)255;
    return p;
  };
  float*  vbar   = (float*)walloc(32 * 768 * 4);
  float*  tbar   = (float*)walloc(32 * 768 * 4);
  bf16_t* patchN = (bf16_t*)walloc((size_t)32 * 192 * 256 * 2);
  bf16_t* patchT = (bf16_t*)walloc((size_t)32 * 256 * 192 * 2);
  bf16_t* attN   = (bf16_t*)walloc((size_t)32 * 32 * 256 * 2);
  bf16_t* attT   = (bf16_t*)walloc((size_t)32 * 256 * 32 * 2);
  // bf16 copies of inputs (converted once; GEMMs then stream bf16 from L2)
  const size_t nVF = (size_t)32 * 193 * 768;
  const size_t nTF = (size_t)32 * 31 * 768;
  const size_t nW  = (size_t)256 * 768;
  const size_t nBar = (size_t)32 * 768;
  bf16_t* vf_bf = (bf16_t*)walloc(nVF * 2);
  bf16_t* tf_bf = (bf16_t*)walloc(nTF * 2);
  bf16_t* Wv_bf = (bf16_t*)walloc(nW * 2);
  bf16_t* Wt_bf = (bf16_t*)walloc(nW * 2);
  bf16_t* Wp_bf = (bf16_t*)walloc(nW * 2);
  bf16_t* Wa_bf = (bf16_t*)walloc(nW * 2);
  bf16_t* vbar_bf = (bf16_t*)walloc(nBar * 2);
  bf16_t* tbar_bf = (bf16_t*)walloc(nBar * 2);

  auto cvt = [&](const float* s, bf16_t* d, size_t n) {
    int n2 = (int)(n / 2);
    cvt_f32_bf16<<<(n2 + 255) / 256, 256, 0, stream>>>(s, (unsigned*)d, n2);
  };

  // zero attN + attT (contiguous) so padded attribute rows are zero
  {
    int nw = (int)(((size_t)32 * 32 * 256 * 2 + (size_t)32 * 256 * 32 * 2) / 4);
    zero_u32<<<(nw + 255) / 256, 256, 0, stream>>>((unsigned*)attN, nw);
  }
  cvt(vf, vf_bf, nVF);
  cvt(tf, tf_bf, nTF);
  cvt(W_vis, Wv_bf, nW);
  cvt(W_txt, Wt_bf, nW);
  cvt(W_patch, Wp_bf, nW);
  cvt(W_att, Wa_bf, nW);

  means_kernel<<<64, 256, 0, stream>>>(vf, tf, vbar, tbar);
  cvt(vbar, vbar_bf, nBar);
  cvt(tbar, tbar_bf, nBar);

  proj_kernel<<<1, 256, 0, stream>>>(vf_bf, Wv_bf, b_vis, att_nums, out_vis, nullptr, nullptr, 0);
  proj_kernel<<<1, 256, 0, stream>>>(tf_bf, Wt_bf, b_txt, att_nums, out_txt, nullptr, nullptr, 1);
  proj_kernel<<<1, 256, 0, stream>>>(vbar_bf, Wp_bf, b_patch, att_nums, out_pm, nullptr, nullptr, 2);
  proj_kernel<<<1, 256, 0, stream>>>(tbar_bf, Wa_bf, b_att, att_nums, out_am, nullptr, nullptr, 3);
  proj_kernel<<<192, 256, 0, stream>>>(vf_bf, Wp_bf, b_patch, att_nums, nullptr, patchN, patchT, 4);
  proj_kernel<<<30, 256, 0, stream>>>(tf_bf, Wa_bf, b_att, att_nums, nullptr, attN, attT, 5);

  size_t smemBytes = (size_t)(32 * 192 + 32 * 256 + 16 * 256 + 256 + 256 + 512 + 32 + 32 + 32) * 4
                   + (size_t)(32 * 192 + 192 * 32) * 2;   // = 102784 B
  pair_kernel<<<1024, 256, smemBytes, stream>>>(patchN, patchT, attN, attT, att_nums, out_sim);
}
